// EdgeConvFeatureExtractor_18537078849895
// MI455X (gfx1250) — matrix operand order repaired and checked by reading, not compiled, not run on hardware
//
#include <hip/hip_runtime.h>

// EdgeConv feature extractor for MI455X (gfx1250, wave32, WMMA).
// Pipeline: knn -> weight transpose/f16 -> edgeconv1 (VALU) -> edgeconv2 (WMMA).

#define P_TOTAL 8192   // B*N = 2*4096
#define KNN     8
#define H1      128
#define OUTC    256

typedef _Float16 v16h __attribute__((ext_vector_type(16)));
typedef _Float16 v8h  __attribute__((ext_vector_type(8)));
typedef float    v8f  __attribute__((ext_vector_type(8)));

static __device__ __forceinline__ v8f wmma_f16(v16h a, v16h b, v8f c) {
  // D = A(16x32 f16) * B(32x16 f16) + C(16x16 f32)
  return __builtin_amdgcn_wmma_f32_16x16x32_f16(
      /*neg_a=*/false, a, /*neg_b=*/false, b,
      /*c_mod=*/(short)0, c, /*reuse_a=*/false, /*reuse_b=*/false);
}

// A-fragment (16x32 f16, ISA 7.12.2 layout): lanes 0-15 row m=lane hold
// K = k0+0..7 (elems 0-7) and K = k0+16..23 (elems 8-15); lanes 16-31 get +8.
static __device__ __forceinline__ v16h ld_a_frag(const _Float16* base, int ldh,
                                                 int rowBase, int k0, int lane) {
  int m  = lane & 15;
  int kb = k0 + ((lane < 16) ? 0 : 8);
  const _Float16* p = base + (rowBase + m) * ldh + kb;
  v8h lo = *(const v8h*)(p);
  v8h hi = *(const v8h*)(p + 16);
  return __builtin_shufflevector(lo, hi, 0,1,2,3,4,5,6,7,8,9,10,11,12,13,14,15);
}

// B-fragment (32x16 f16): lane holds column n=lane&15; lanes 0-15 hold
// K = k0+0..7 and k0+16..23, lanes 16-31 get +8. Weights staged transposed
// ([n][k]) in LDS so K is contiguous -> two 16B loads.
static __device__ __forceinline__ v16h ld_b_frag(const _Float16* baseT, int ldh,
                                                 int n0, int k0, int lane) {
  int n  = lane & 15;
  int kb = k0 + ((lane < 16) ? 0 : 8);
  const _Float16* p = baseT + (n0 + n) * ldh + kb;
  v8h lo = *(const v8h*)(p);
  v8h hi = *(const v8h*)(p + 16);
  return __builtin_shufflevector(lo, hi, 0,1,2,3,4,5,6,7,8,9,10,11,12,13,14,15);
}

// ---------------------------------------------------------------- KNN (K=8)
__global__ void knn_kernel(const float* __restrict__ pts, int* __restrict__ idx) {
  __shared__ float tx[256], ty[256], tz[256];
  const int tid = threadIdx.x;
  const int i = blockIdx.x * 256 + tid;
  const float xi = pts[i * 3 + 0], yi = pts[i * 3 + 1], zi = pts[i * 3 + 2];

  float bd[KNN];
  int   bi[KNN];
#pragma unroll
  for (int q = 0; q < KNN; ++q) { bd[q] = 3.4e38f; bi[q] = 0; }

  for (int t = 0; t < P_TOTAL / 256; ++t) {
    const int j0 = t * 256;
    tx[tid] = pts[(j0 + tid) * 3 + 0];
    ty[tid] = pts[(j0 + tid) * 3 + 1];
    tz[tid] = pts[(j0 + tid) * 3 + 2];
    __syncthreads();
    if (t + 1 < P_TOTAL / 256)  // pull next tile toward L2/WGP$ (global_prefetch_b8)
      __builtin_prefetch((const void*)(pts + (j0 + 256 + tid) * 3), 0, 1);
    for (int jj = 0; jj < 256; ++jj) {
      const float dx = xi - tx[jj], dy = yi - ty[jj], dz = zi - tz[jj];
      const float d = dx * dx + dy * dy + dz * dz;
      const int   j = j0 + jj;
      if (d < bd[KNN - 1]) {
        // register-resident sorted insert; ties keep earlier index (top_k-like)
#pragma unroll
        for (int q = KNN - 1; q >= 0; --q) {
          const bool shift = (q > 0) && (d < bd[q - 1]);
          const bool place = (d < bd[q]) && !shift;
          const float pv = shift ? bd[q - 1] : (place ? d : bd[q]);
          const int   iv = shift ? bi[q - 1] : (place ? j : bi[q]);
          bd[q] = pv; bi[q] = iv;
        }
      }
    }
    __syncthreads();
  }
#pragma unroll
  for (int q = 0; q < KNN; ++q) idx[i * KNN + q] = bi[q];
}

// ------------------------------------- transpose + f16-convert W3, W4 for WMMA B
__global__ void prep_weights(const float* __restrict__ W3, const float* __restrict__ W4,
                             _Float16* __restrict__ W3T, _Float16* __restrict__ W4T) {
  const int t = blockIdx.x * 256 + threadIdx.x;  // 65536 threads
  if (t < 32768) {                 // W3: [256][128] row-major -> W3T[n][k] (128x256)
    const int k = t >> 7, n = t & 127;
    W3T[n * 256 + k] = (_Float16)W3[t];
  } else {                         // W4: [128][256] row-major -> W4T[n][k] (256x128)
    const int u = t - 32768;
    const int k = u >> 8, n = u & 255;
    W4T[n * 128 + k] = (_Float16)W4[u];
  }
}

// ---------------------------------------------- EdgeConv1: 6 -> 64 -> 128 (VALU)
// K-dims too small for WMMA (6, 64); keep weights LDS-resident, one wave/point.
__global__ void edgeconv1_kernel(const float* __restrict__ pts, const int* __restrict__ idx,
                                 const float* __restrict__ W1, const float* __restrict__ b1,
                                 const float* __restrict__ W2, const float* __restrict__ b2,
                                 _Float16* __restrict__ x1h) {
  __shared__ float sW1[6 * 64];
  __shared__ float sB1[64];
  __shared__ float sW2[64 * 128];
  __shared__ float sB2[128];
  __shared__ float sh0[8][64];

  const int tid = threadIdx.x;
  for (int o = tid; o < 6 * 64; o += 256) sW1[o] = W1[o];
  for (int o = tid; o < 64 * 128; o += 256) sW2[o] = W2[o];
  if (tid < 64)  sB1[tid] = b1[tid];
  if (tid < 128) sB2[tid] = b2[tid];
  __syncthreads();

  const int w = tid >> 5, lane = tid & 31;
  const int i = blockIdx.x * 8 + w;
  const float xi0 = pts[i * 3 + 0], xi1 = pts[i * 3 + 1], xi2 = pts[i * 3 + 2];

  float mx[4] = {-3.4e38f, -3.4e38f, -3.4e38f, -3.4e38f};

  for (int nb = 0; nb < KNN; ++nb) {
    const int j = idx[i * KNN + nb];
    const float e3 = pts[j * 3 + 0] - xi0;
    const float e4 = pts[j * 3 + 1] - xi1;
    const float e5 = pts[j * 3 + 2] - xi2;
#pragma unroll
    for (int q = 0; q < 2; ++q) {
      const int c0 = lane + 32 * q;
      float s = sB1[c0];
      s += xi0 * sW1[0 * 64 + c0] + xi1 * sW1[1 * 64 + c0] + xi2 * sW1[2 * 64 + c0];
      s += e3 * sW1[3 * 64 + c0] + e4 * sW1[4 * 64 + c0] + e5 * sW1[5 * 64 + c0];
      sh0[w][c0] = fmaxf(s, 0.f);
    }
    __syncthreads();
#pragma unroll
    for (int q = 0; q < 4; ++q) {
      const int c = lane + 32 * q;
      float s = sB2[c];
      for (int c0 = 0; c0 < 64; ++c0) s += sh0[w][c0] * sW2[c0 * 128 + c];
      mx[q] = fmaxf(mx[q], s);
    }
    __syncthreads();
  }
#pragma unroll
  for (int q = 0; q < 4; ++q)
    x1h[i * H1 + lane + 32 * q] = (_Float16)fmaxf(mx[q], 0.f);  // outer ReLU
}

// ------------------------------- EdgeConv2: 256 -> 128 -> 256 with v_wmma f16
// One workgroup = 16 points = 128 edge rows. 8 waves, each owns a 16-row M tile.
// A fragments are register-cached across the whole n-loop (wave's M tile is
// n-invariant), so steady state is 2 ds_load_b128 (B) per v_wmma.
#define LDE  264   // sE row stride (256 + 8 pad halves)
#define LDW1 264   // staged W3T row stride
#define LDH  136   // sH row stride (128 + 8)
#define LDW2 136   // staged W4T row stride
#define OFF_SW 67584    // 128*264*2
#define OFF_SH 137216   // OFF_SW + 256*136*2
#define SMEM_BYTES 172032

__global__ void edgeconv2_kernel(const _Float16* __restrict__ x1h, const int* __restrict__ idx,
                                 const _Float16* __restrict__ W3T, const float* __restrict__ b3,
                                 const _Float16* __restrict__ W4T, const float* __restrict__ b4,
                                 float* __restrict__ out) {
  extern __shared__ char smem[];
  _Float16* sE = (_Float16*)smem;              // 128 x 256 edge features
  _Float16* sW = (_Float16*)(smem + OFF_SW);   // reused weight buffer
  _Float16* sH = (_Float16*)(smem + OFF_SH);   // 128 x 128 hidden

  const int tid = threadIdx.x;
  const int wave = tid >> 5, lane = tid & 31;
  const int pBase = blockIdx.x * 16;

  // Build edge rows: e = [x_i, x_j - x_i]; thread t -> row t/2, half t&1.
  {
    const int row = tid >> 1, half = tid & 1;
    const int ip = pBase + (row >> 3);
    const _Float16* xi = x1h + ip * H1;
    _Float16* dst = sE + row * LDE + half * H1;
    if (half == 0) {
#pragma unroll
      for (int v = 0; v < 16; ++v)
        *(v8h*)(dst + v * 8) = *(const v8h*)(xi + v * 8);
    } else {
      const int j = idx[ip * KNN + (row & 7)];
      const _Float16* xj = x1h + j * H1;
#pragma unroll
      for (int v = 0; v < 16; ++v) {
        v8h a = *(const v8h*)(xi + v * 8);
        v8h b = *(const v8h*)(xj + v * 8);
        *(v8h*)(dst + v * 8) = b - a;
      }
    }
  }
  // Stage W3T (128 x 256) into padded sW.
  {
    const int row = tid >> 1, half = tid & 1;
    const _Float16* src = W3T + row * 256 + half * 128;
    _Float16* dst = sW + row * LDW1 + half * 128;
#pragma unroll
    for (int v = 0; v < 16; ++v)
      *(v8h*)(dst + v * 8) = *(const v8h*)(src + v * 8);
  }
  __syncthreads();

  const int rowBase = wave * 16;
  const int nlane = lane & 15;

  // GEMM1: (128x256) @ W3 (256x128) + b3, ReLU -> sH (f16). N=8 tiles, K=8 steps.
  {
    v16h aF[8];
#pragma unroll
    for (int kt = 0; kt < 8; ++kt)
      aF[kt] = ld_a_frag(sE, LDE, rowBase, kt * 32, lane);

    float bv[8];  // per-lane bias, hoisted off the epilogue critical path
#pragma unroll
    for (int nt = 0; nt < 8; ++nt) bv[nt] = b3[nt * 16 + nlane];

    const int mBase = rowBase + ((lane < 16) ? 0 : 8);
    for (int nt = 0; nt < 8; ++nt) {
      v8f acc = {0.f, 0.f, 0.f, 0.f, 0.f, 0.f, 0.f, 0.f};
#pragma unroll
      for (int kt = 0; kt < 8; ++kt) {
        v16h b = ld_b_frag(sW, LDW1, nt * 16, kt * 32, lane);
        acc = wmma_f16(aF[kt], b, acc);
      }
      const int n = nt * 16 + nlane;
#pragma unroll
      for (int r = 0; r < 8; ++r)
        sH[(mBase + r) * LDH + n] = (_Float16)fmaxf(acc[r] + bv[nt], 0.f);
    }
  }
  __syncthreads();

  // Stage W4T (256 x 128) into padded sW (overwrite).
  {
    const _Float16* src = W4T + tid * 128;
    _Float16* dst = sW + tid * LDW2;
#pragma unroll
    for (int v = 0; v < 16; ++v)
      *(v8h*)(dst + v * 8) = *(const v8h*)(src + v * 8);
  }
  __syncthreads();

  // GEMM2: (128x128) @ W4 (128x256) + b4, then max over each point's 8 edges.
  // D layout: lanes 0-15 hold M=0..7 (point 2w), lanes 16-31 M=8..15 (point 2w+1):
  // neighbor-max is a pure per-lane reduction over the 8 accumulator VGPRs.
  {
    v16h aF[4];
#pragma unroll
    for (int kt = 0; kt < 4; ++kt)
      aF[kt] = ld_a_frag(sH, LDH, rowBase, kt * 32, lane);

    float bv[16];
#pragma unroll
    for (int nt = 0; nt < 16; ++nt) bv[nt] = b4[nt * 16 + nlane];

    const int lp = 2 * wave + ((lane < 16) ? 0 : 1);
    float* outp = out + (pBase + lp) * OUTC;
    for (int nt = 0; nt < 16; ++nt) {
      v8f acc = {0.f, 0.f, 0.f, 0.f, 0.f, 0.f, 0.f, 0.f};
#pragma unroll
      for (int kt = 0; kt < 4; ++kt) {
        v16h b = ld_b_frag(sW, LDW2, nt * 16, kt * 32, lane);
        acc = wmma_f16(aF[kt], b, acc);
      }
      float m = acc[0];
#pragma unroll
      for (int r = 1; r < 8; ++r) m = fmaxf(m, acc[r]);
      outp[nt * 16 + nlane] = m + bv[nt];
    }
  }
}

// ---------------------------------------------------------------- launcher
extern "C" void kernel_launch(void* const* d_in, const int* in_sizes, int n_in,
                              void* d_out, int out_size, void* d_ws, size_t ws_size,
                              hipStream_t stream) {
  const float* pts = (const float*)d_in[0];
  const float* W1  = (const float*)d_in[1];
  const float* b1  = (const float*)d_in[2];
  const float* W2  = (const float*)d_in[3];
  const float* b2  = (const float*)d_in[4];
  const float* W3  = (const float*)d_in[5];
  const float* b3  = (const float*)d_in[6];
  const float* W4  = (const float*)d_in[7];
  const float* b4  = (const float*)d_in[8];
  float* out = (float*)d_out;

  char* ws = (char*)d_ws;
  int*      idx = (int*)ws;                               // 8192*8*4   = 256 KB
  _Float16* x1h = (_Float16*)(ws + 262144);               // 8192*128*2 = 2 MB
  _Float16* W3T = (_Float16*)(ws + 262144 + 2097152);     // 64 KB
  _Float16* W4T = (_Float16*)(ws + 262144 + 2097152 + 65536);  // 64 KB

  knn_kernel<<<P_TOTAL / 256, 256, 0, stream>>>(pts, idx);
  prep_weights<<<65536 / 256, 256, 0, stream>>>(W3, W4, W3T, W4T);
  edgeconv1_kernel<<<P_TOTAL / 8, 256, 0, stream>>>(pts, idx, W1, b1, W2, b2, x1h);
  edgeconv2_kernel<<<P_TOTAL / 16, 256, SMEM_BYTES, stream>>>(x1h, idx, W3T, b3, W4T, b4, out);
}
